// TwinGCL_SimSiam_24017457119851
// MI455X (gfx1250) — matrix operand order repaired and checked by reading, not compiled, not run on hardware
//
#include <hip/hip_runtime.h>

typedef __attribute__((ext_vector_type(16))) _Float16 v16h;
typedef __attribute__((ext_vector_type(8)))  _Float16 v8h;
typedef __attribute__((ext_vector_type(8)))  float    v8f;

// ---------------------------------------------------------------- elementwise
__global__ void k_fill(float* __restrict__ p, float v, int n) {
  int i = blockIdx.x * blockDim.x + threadIdx.x;
  if (i < n) p[i] = v;
}

__global__ void k_deg_acc(const int* __restrict__ dst, float* __restrict__ deg, int E) {
  int i = blockIdx.x * blockDim.x + threadIdx.x;
  if (i < E) atomicAdd(&deg[dst[i]], 1.0f);
}

__global__ void k_rsqrt(const float* __restrict__ deg, float* __restrict__ dinv, int n) {
  int i = blockIdx.x * blockDim.x + threadIdx.x;
  if (i < n) dinv[i] = rsqrtf(deg[i]);
}

__global__ void k_f32_to_f16(const float* __restrict__ x, _Float16* __restrict__ y, int n) {
  int i = blockIdx.x * blockDim.x + threadIdx.x;
  if (i < n) y[i] = (_Float16)x[i];
}

// Pack W[K,Nc] (row-major f32) into WMMA-ready f16 tiles:
// Wp[(((ks*nbTot+nbG)*32+lane)*16)+j] = W[ks*32 + (j&7) + 8*(lane>>4) + 16*(j>>3), nbG*16 + (lane&15)]
__global__ void k_pack_w(const float* __restrict__ W, _Float16* __restrict__ Wp,
                         int K, int Nc, int total) {
  int p = blockIdx.x * blockDim.x + threadIdx.x;
  if (p >= total) return;
  int j    = p & 15;
  int lane = (p >> 4) & 31;
  int nbTot = Nc >> 4;
  int rest = p >> 9;
  int nbG  = rest % nbTot;
  int ks   = rest / nbTot;
  int k = ks * 32 + (j & 7) + ((lane >> 4) << 3) + ((j >> 3) << 4);
  int n = (nbG << 4) + (lane & 15);
  Wp[p] = (_Float16)W[(size_t)k * Nc + n];
}

// ---------------------------------------------------------------- WMMA GEMM
// C[M,Nc] = A[M,K] @ W (packed). One wave => 16 rows x 128 cols.
// M % 16 == 0, K % 32 == 0, Nc % 128 == 0.
// B fragments for all 8 column blocks are fetched in one load clause per
// K-step so each WMMA only waits on its own fragment (partial loadcnt waits).
__global__ void k_gemm_wmma(const _Float16* __restrict__ A,
                            const _Float16* __restrict__ Wp,
                            float* __restrict__ C,
                            int M, int K, int Nc) {
  const int lane = threadIdx.x & 31;
  const int wave = (blockIdx.x * blockDim.x + threadIdx.x) >> 5;
  const int colWaves = Nc >> 7;
  const int totalWaves = (M >> 4) * colWaves;
  if (wave >= totalWaves) return;               // wave-uniform: EXEC all-1 below
  const int rowTile = wave / colWaves;
  const int colWave = wave - rowTile * colWaves;
  const int half_ = lane >> 4;
  const int row   = (rowTile << 4) + (lane & 15);
  const int nbTot = Nc >> 4;
  const int kSteps = K >> 5;

  // A fragment base: 16-bit 16x32 layout (lanes 0-15: K 0-7 & 16-23; +8 for hi lanes)
  const _Float16* aptr = A + (size_t)row * K + half_ * 8;
  // W fragment base for this wave's first column block; nb steps by 512 elements
  // (1024 B -> folds into the 24-bit immediate offset), ks steps by wKstride.
  const _Float16* wptr = Wp + ((size_t)(colWave << 3) * 32 + lane) * 16;
  const size_t wKstride = (size_t)nbTot * 32 * 16;

  v8f acc[8];
#pragma unroll
  for (int i = 0; i < 8; ++i) acc[i] = (v8f)(0.0f);

  for (int ks = 0; ks < kSteps; ++ks) {
    v8h a0 = *(const v8h*)(aptr);
    v8h a1 = *(const v8h*)(aptr + 16);
    aptr += 32;
    v16h b[8];
#pragma unroll
    for (int nb = 0; nb < 8; ++nb)
      b[nb] = *(const v16h*)(wptr + nb * (32 * 16));
    wptr += wKstride;
    v16h a;
#pragma unroll
    for (int j = 0; j < 8; ++j) { a[j] = a0[j]; a[8 + j] = a1[j]; }
#pragma unroll
    for (int nb = 0; nb < 8; ++nb)
      acc[nb] = __builtin_amdgcn_wmma_f32_16x16x32_f16(
          false, a, false, b[nb], (short)0, acc[nb], false, false);
  }
  // D layout: VGPR g -> row = g + 8*half, col = lane&15
  const int n0 = (colWave << 7) + (lane & 15);
  const int rbase = (rowTile << 4) + half_ * 8;
  float* cptr = C + (size_t)rbase * Nc + n0;
#pragma unroll
  for (int g = 0; g < 8; ++g) {
#pragma unroll
    for (int nb = 0; nb < 8; ++nb) cptr[nb * 16] = acc[nb][g];
    cptr += Nc;
  }
}

// ---------------------------------------------------------------- GCN pieces
__global__ void k_init_agg(const float* __restrict__ t, const float* __restrict__ dinv,
                           float* __restrict__ agg, int total) {
  int i = blockIdx.x * blockDim.x + threadIdx.x;
  if (i >= total) return;
  int n = i >> 7;
  float d = dinv[n];
  agg[i] = d * d * t[i];
}

__global__ void k_scatter(const int* __restrict__ src, const int* __restrict__ dst,
                          const float* __restrict__ dinv, const float* __restrict__ t,
                          float* __restrict__ agg, int E) {
  int wave = (blockIdx.x * blockDim.x + threadIdx.x) >> 5;
  int lane = threadIdx.x & 31;
  if (wave >= E) return;
  int s = src[wave], d = dst[wave];
  float nrm = dinv[s] * dinv[d];
  float4 v = ((const float4*)(t + (size_t)s * 128))[lane];
  float* ad = agg + (size_t)d * 128 + lane * 4;
  atomicAdd(ad + 0, nrm * v.x);
  atomicAdd(ad + 1, nrm * v.y);
  atomicAdd(ad + 2, nrm * v.z);
  atomicAdd(ad + 3, nrm * v.w);
}

__global__ void k_epilogue(const float* __restrict__ agg, const float* __restrict__ b,
                           const float* __restrict__ res, float* __restrict__ hout,
                           _Float16* __restrict__ hh, int total) {
  int i = blockIdx.x * blockDim.x + threadIdx.x;
  if (i >= total) return;
  float v = fmaxf(agg[i] + b[i & 127], 0.0f);
  if (res) v += res[i];
  hout[i] = v;
  hh[i] = (_Float16)v;
}

__global__ void k_pool_acc(const float* __restrict__ h, const int* __restrict__ batch,
                           float* __restrict__ pooled, float* __restrict__ cnt, int N) {
  int wave = (blockIdx.x * blockDim.x + threadIdx.x) >> 5;
  int lane = threadIdx.x & 31;
  if (wave >= N) return;
  int g = batch[wave];
  float4 v = ((const float4*)(h + (size_t)wave * 128))[lane];
  float* pd = pooled + (size_t)g * 128 + lane * 4;
  atomicAdd(pd + 0, v.x);
  atomicAdd(pd + 1, v.y);
  atomicAdd(pd + 2, v.z);
  atomicAdd(pd + 3, v.w);
  if (lane == 0) atomicAdd(cnt + g, 1.0f);
}

__global__ void k_pool_div(float* __restrict__ pooled, const float* __restrict__ cnt,
                           int total, int C) {
  int i = blockIdx.x * blockDim.x + threadIdx.x;
  if (i < total) pooled[i] /= fmaxf(cnt[i / C], 1.0f);
}

// ---------------------------------------------------------------- heads
// y = gamma*((x+bias)-mean)/sqrt(var+eps)+beta  (biased var over rows), opt ReLU
__global__ void k_bn(const float* __restrict__ x, const float* __restrict__ bias,
                     const float* __restrict__ gamma, const float* __restrict__ beta,
                     float* __restrict__ y, int rows, int cols, int relu) {
  __shared__ float s0[128], s1[128];
  int col = blockIdx.x;
  int tid = threadIdx.x;
  float b = bias[col];
  float sum = 0.f, sq = 0.f;
  for (int r = tid; r < rows; r += 128) {
    float v = x[(size_t)r * cols + col] + b;
    sum += v; sq += v * v;
  }
  s0[tid] = sum; s1[tid] = sq;
  __syncthreads();
  for (int off = 64; off > 0; off >>= 1) {
    if (tid < off) { s0[tid] += s0[tid + off]; s1[tid] += s1[tid + off]; }
    __syncthreads();
  }
  float mean = s0[0] / rows;
  float var  = s1[0] / rows - mean * mean;
  float inv  = rsqrtf(var + 1e-5f);
  float g = gamma[col], be = beta[col];
  for (int r = tid; r < rows; r += 128) {
    float v = (x[(size_t)r * cols + col] + b - mean) * inv * g + be;
    if (relu) v = fmaxf(v, 0.f);
    y[(size_t)r * cols + col] = v;
  }
}

__global__ void k_bias_act(const float* __restrict__ x, const float* __restrict__ b,
                           float* __restrict__ y, int total, int cols, int relu) {
  int i = blockIdx.x * blockDim.x + threadIdx.x;
  if (i >= total) return;
  float v = x[i] + b[i % cols];
  if (relu) v = fmaxf(v, 0.f);
  y[i] = v;
}

// ---------------------------------------------------------------- host
static inline int cdiv(int a, int b) { return (a + b - 1) / b; }

extern "C" void kernel_launch(void* const* d_in, const int* in_sizes, int n_in,
                              void* d_out, int out_size, void* d_ws, size_t ws_size,
                              hipStream_t stream) {
  const float* x1  = (const float*)d_in[0];
  const int* ei1   = (const int*)d_in[1];
  const int* bat1  = (const int*)d_in[2];
  const float* x2  = (const float*)d_in[3];
  const int* ei2   = (const int*)d_in[4];
  const int* bat2  = (const int*)d_in[5];
  const float* W1  = (const float*)d_in[6];  const float* b1  = (const float*)d_in[7];
  const float* W2  = (const float*)d_in[8];  const float* b2  = (const float*)d_in[9];
  const float* W3  = (const float*)d_in[10]; const float* b3  = (const float*)d_in[11];
  const float* Wp1 = (const float*)d_in[12]; const float* bp1 = (const float*)d_in[13];
  const float* g1  = (const float*)d_in[14]; const float* be1 = (const float*)d_in[15];
  const float* Wp2 = (const float*)d_in[16]; const float* bp2 = (const float*)d_in[17];
  const float* g2  = (const float*)d_in[18]; const float* be2 = (const float*)d_in[19];
  const float* Wq1 = (const float*)d_in[20]; const float* bq1 = (const float*)d_in[21];
  const float* Wq2 = (const float*)d_in[22]; const float* bq2 = (const float*)d_in[23];

  const int N  = in_sizes[0] / 128;   // 50000
  const int E  = in_sizes[1] / 2;     // 1600000
  const int Gn = 512;
  const int NF = N * 128;

  // bump allocator over workspace (256B aligned)
  char* ws = (char*)d_ws;
  size_t off = 0;
  auto alloc = [&](size_t bytes) -> void* {
    void* p = ws + off;
    off = (off + bytes + 255) & ~(size_t)255;
    return p;
  };
  _Float16* Xh  = (_Float16*)alloc((size_t)NF * 2);
  float* T      = (float*)alloc((size_t)NF * 4);
  float* Agg    = (float*)alloc((size_t)NF * 4);
  float* Ha     = (float*)alloc((size_t)NF * 4);
  float* Hb     = (float*)alloc((size_t)NF * 4);
  float* deg    = (float*)alloc((size_t)N * 4);
  float* dinv   = (float*)alloc((size_t)N * 4);
  float* pooled = (float*)alloc((size_t)Gn * 128 * 4);
  float* cnt    = (float*)alloc((size_t)Gn * 4);
  _Float16* Sh  = (_Float16*)alloc((size_t)Gn * 256 * 2);
  float* SA     = (float*)alloc((size_t)Gn * 256 * 4);
  float* SB     = (float*)alloc((size_t)Gn * 256 * 4);
  _Float16* W1p  = (_Float16*)alloc(128 * 128 * 2);
  _Float16* W2p  = (_Float16*)alloc(128 * 128 * 2);
  _Float16* W3p  = (_Float16*)alloc(128 * 128 * 2);
  _Float16* Wp1p = (_Float16*)alloc(128 * 256 * 2);
  _Float16* Wp2p = (_Float16*)alloc(256 * 256 * 2);
  _Float16* Wq1p = (_Float16*)alloc(256 * 128 * 2);
  _Float16* Wq2p = (_Float16*)alloc(128 * 256 * 2);
  (void)ws_size; (void)n_in; (void)out_size;

  // pack all weights into WMMA-ready f16 layout (re-done every call; cheap)
  auto pack = [&](const float* W, _Float16* Wp, int K, int Nc) {
    int tot = K * Nc;
    k_pack_w<<<cdiv(tot, 256), 256, 0, stream>>>(W, Wp, K, Nc, tot);
  };
  pack(W1, W1p, 128, 128);  pack(W2, W2p, 128, 128);  pack(W3, W3p, 128, 128);
  pack(Wp1, Wp1p, 128, 256); pack(Wp2, Wp2p, 256, 256);
  pack(Wq1, Wq1p, 256, 128); pack(Wq2, Wq2p, 128, 256);

  auto gemm = [&](const _Float16* A, const _Float16* Wp, float* C, int M, int K, int Nc) {
    int waves = (M / 16) * (Nc / 128);
    k_gemm_wmma<<<cdiv(waves * 32, 256), 256, 0, stream>>>(A, Wp, C, M, K, Nc);
  };

  float* out = (float*)d_out;
  float* p1o = out;
  float* p2o = out + 512 * 256;
  float* z1o = out + 2 * 512 * 256;
  float* z2o = out + 3 * 512 * 256;

  auto run_view = [&](const float* x, const int* ei, const int* batch,
                      float* z_out, float* p_out) {
    const int* src = ei;
    const int* dst = ei + E;
    // symmetric-norm degrees (self-loop included)
    k_fill<<<cdiv(N, 256), 256, 0, stream>>>(deg, 1.0f, N);
    k_deg_acc<<<cdiv(E, 256), 256, 0, stream>>>(dst, deg, E);
    k_rsqrt<<<cdiv(N, 256), 256, 0, stream>>>(deg, dinv, N);
    // layer input
    k_f32_to_f16<<<cdiv(NF, 256), 256, 0, stream>>>(x, Xh, NF);

    const _Float16* Wps[3] = {W1p, W2p, W3p};
    const float* bs[3]     = {b1, b2, b3};
    float* outs[3]         = {Ha, Hb, Ha};
    const float* ress[3]   = {nullptr, Ha, Hb};
    for (int l = 0; l < 3; ++l) {
      gemm(Xh, Wps[l], T, N, 128, 128);                                    // t = h @ W
      k_init_agg<<<cdiv(NF, 256), 256, 0, stream>>>(T, dinv, Agg, NF);     // self-loop
      k_scatter<<<cdiv(E, 8), 256, 0, stream>>>(src, dst, dinv, T, Agg, E);
      k_epilogue<<<cdiv(NF, 256), 256, 0, stream>>>(Agg, bs[l], ress[l], outs[l], Xh, NF);
    }
    // mean pool (h3 lives in Ha)
    k_fill<<<cdiv(Gn * 128, 256), 256, 0, stream>>>(pooled, 0.f, Gn * 128);
    k_fill<<<cdiv(Gn, 256), 256, 0, stream>>>(cnt, 0.f, Gn);
    k_pool_acc<<<cdiv(N, 8), 256, 0, stream>>>(Ha, batch, pooled, cnt, N);
    k_pool_div<<<cdiv(Gn * 128, 256), 256, 0, stream>>>(pooled, cnt, Gn * 128, 128);
    // projector
    k_f32_to_f16<<<cdiv(Gn * 128, 256), 256, 0, stream>>>(pooled, Sh, Gn * 128);
    gemm(Sh, Wp1p, SA, Gn, 128, 256);
    k_bn<<<256, 128, 0, stream>>>(SA, bp1, g1, be1, SB, Gn, 256, 1);
    k_f32_to_f16<<<cdiv(Gn * 256, 256), 256, 0, stream>>>(SB, Sh, Gn * 256);
    gemm(Sh, Wp2p, SA, Gn, 256, 256);
    k_bn<<<256, 128, 0, stream>>>(SA, bp2, g2, be2, z_out, Gn, 256, 0);
    // predictor
    k_f32_to_f16<<<cdiv(Gn * 256, 256), 256, 0, stream>>>(z_out, Sh, Gn * 256);
    gemm(Sh, Wq1p, SA, Gn, 256, 128);
    k_bias_act<<<cdiv(Gn * 128, 256), 256, 0, stream>>>(SA, bq1, SB, Gn * 128, 128, 1);
    k_f32_to_f16<<<cdiv(Gn * 128, 256), 256, 0, stream>>>(SB, Sh, Gn * 128);
    gemm(Sh, Wq2p, SA, Gn, 128, 256);
    k_bias_act<<<cdiv(Gn * 256, 256), 256, 0, stream>>>(SA, bq2, p_out, Gn * 256, 256, 0);
  };

  run_view(x1, ei1, bat1, z1o, p1o);
  run_view(x2, ei2, bat2, z2o, p2o);
}
